// SingleHeadCausalSelfAttention_33895881900335
// MI455X (gfx1250) — compile-verified, matrix operand order checked by reading, not run on hardware
//
#include <hip/hip_runtime.h>
#include <hip/hip_bf16.h>

typedef __attribute__((ext_vector_type(16))) __bf16 v16bf;
typedef __attribute__((ext_vector_type(8)))  __bf16 v8bf;
typedef __attribute__((ext_vector_type(8)))  float  v8f;

#define B_  4
#define S_  4096
#define D_  1024
#define HD_ 64

// K index inside a 16-bit A/B operand: VGPR j (0..7), lane group g (0..1).
__device__ __forceinline__ int kb_of(int j, int g) {
    return ((j & 3) << 1) + (g << 3) + ((j >> 2) << 4);
}

__device__ __forceinline__ v8f wmma_bf16(v16bf a, v16bf b, v8f c) {
    return __builtin_amdgcn_wmma_f32_16x16x32_bf16(
        /*neg_a=*/false, a, /*neg_b=*/false, b,
        /*c_mod=*/(short)0, c, /*reuse_a=*/false, /*reuse_b=*/false);
}

__device__ __forceinline__ v8f v8f_zero() {
    v8f z = {0.f, 0.f, 0.f, 0.f, 0.f, 0.f, 0.f, 0.f};
    return z;
}

__device__ __forceinline__ v16bf cat8(v8bf lo, v8bf hi) {
    return __builtin_shufflevector(lo, hi, 0, 1, 2, 3, 4, 5, 6, 7,
                                           8, 9, 10, 11, 12, 13, 14, 15);
}

// 16-bit A/B operand covering K = 0..31 of `base`: per lane two contiguous
// 16-byte chunks at +8g and +16+8g (operand elements 0..7 / 8..15).
__device__ __forceinline__ v16bf load_op16(const __bf16* base, int g) {
    v8bf c0 = *(const v8bf*)(base + 8 * g);
    v8bf c1 = *(const v8bf*)(base + 16 + 8 * g);
    return cat8(c0, c1);
}

// Same, but from f32 memory with on-the-fly bf16 conversion (A operand of
// the projection — the one conversion that must stream with x).
__device__ __forceinline__ v16bf load_cvt16(const float* base, int g) {
    v8f f0 = *(const v8f*)(base + 8 * g);
    v8f f1 = *(const v8f*)(base + 16 + 8 * g);
    v8bf c0 = __builtin_convertvector(f0, v8bf);
    v8bf c1 = __builtin_convertvector(f1, v8bf);
    return cat8(c0, c1);
}

// wsw layout: [mat][kk/32][ntile][lane][16] bf16 — each lane's 16 operand
// elements contiguous (32 B) in exact B-operand order.
__device__ __forceinline__ size_t widx(int mat, int kk32, int t, int lane) {
    return ((((size_t)mat * (D_ / 32) + kk32) * 4 + t) * 32 + lane) * 16;
}

// ---------------------------------------------------------------------------
// Kernel 0: one-shot weight convert + swizzle into B-operand order.
// ---------------------------------------------------------------------------
__global__ void wswizzle_kernel(const float* __restrict__ wq,
                                const float* __restrict__ wk,
                                const float* __restrict__ wv,
                                __bf16* __restrict__ wsw)
{
    const int idx = blockIdx.x * blockDim.x + threadIdx.x;  // 196608 total
    const int e    = idx & 15;
    const int lane = (idx >> 4) & 31;
    const int t    = (idx >> 9) & 3;
    const int kk32 = (idx >> 11) & 31;
    const int mat  = idx >> 16;
    const int g = lane >> 4;
    const int n = t * 16 + (lane & 15);
    const int K = kk32 * 32 + kb_of(e >> 1, g) + (e & 1);
    const float* w = (mat == 0) ? wq : (mat == 1) ? wk : wv;
    wsw[idx] = (__bf16)w[(size_t)n * D_ + K];
}

// ---------------------------------------------------------------------------
// Kernel 1: fused QKV projection.  One wave owns a 16-row tile of x, converts
// the A operand ONCE per K-step, and accumulates q, k, v (12 WMMA per K-step)
// with zero-VALU pre-swizzled bf16 weight operands.  x streamed exactly once.
// __launch_bounds__(256, 1): 12 persistent accumulator tiles (96 VGPRs) plus
// deep load pipelining need the full per-wave VGPR budget — without it the RA
// spilled an accumulator to scratch every K-step.
// ---------------------------------------------------------------------------
__global__ __launch_bounds__(256, 1)
void qkv_proj_kernel(const float* __restrict__ x,
                     const __bf16* __restrict__ wsw,
                     const float* __restrict__ bq,
                     const float* __restrict__ bk,
                     const float* __restrict__ bv,
                     __bf16* __restrict__ qo, __bf16* __restrict__ ko,
                     __bf16* __restrict__ vto)
{
    const int lane = threadIdx.x & 31;
    const int wid  = (blockIdx.x * blockDim.x + threadIdx.x) >> 5;
    const int b    = wid / (S_ / 16);
    const int s0   = (wid % (S_ / 16)) * 16;

    const int m = lane & 15;
    const int g = lane >> 4;

    const float* bvec[3] = {bq, bk, bv};

    v8f acc[3][4];
    #pragma unroll
    for (int mt = 0; mt < 3; ++mt)
        #pragma unroll
        for (int t = 0; t < 4; ++t) acc[mt][t] = v8f_zero();

    const float* xrow = x + (size_t)(b * S_ + s0 + m) * D_;

    for (int kk = 0; kk < D_; kk += 32) {
        const v16bf a = load_cvt16(xrow + kk, g);   // shared by 12 WMMAs
        const int kk32 = kk >> 5;
        #pragma unroll
        for (int mt = 0; mt < 3; ++mt) {
            #pragma unroll
            for (int t = 0; t < 4; ++t) {
                const v16bf bm = *(const v16bf*)(wsw + widx(mt, kk32, t, lane));
                acc[mt][t] = wmma_bf16(a, bm, acc[mt][t]);
            }
        }
    }

    const float qscale = 0.125f;  // 1/sqrt(64)
    #pragma unroll
    for (int mt = 0; mt < 3; ++mt) {
        #pragma unroll
        for (int t = 0; t < 4; ++t) {
            const int n = t * 16 + m;
            const float bn = bvec[mt][n];
            #pragma unroll
            for (int r = 0; r < 8; ++r) {
                const int row = r + 8 * g;
                const float val = acc[mt][t][r] + bn;
                if (mt == 0) {
                    qo[(size_t)(b * S_ + s0 + row) * HD_ + n] = (__bf16)(val * qscale);
                } else if (mt == 1) {
                    ko[(size_t)(b * S_ + s0 + row) * HD_ + n] = (__bf16)val;
                } else {
                    vto[(size_t)(b * HD_ + n) * S_ + (s0 + row)] = (__bf16)val;
                }
            }
        }
    }
}

// ---------------------------------------------------------------------------
// Kernel 2: causal flash attention, 32 query rows per wave (two 16-row
// halves h=0,1 share every K/V tile -> 16 WMMA per 32-key block for one set
// of K/V loads).  Online softmax in f32; P tile bounces through per-wave LDS
// (same-wave DS ops are in-order per the ISA -> no barrier, safe with
// divergent causal trip counts).  Next block's K/V rows are prefetched.
// __launch_bounds__(128, 1): ~220 live VGPRs (O + Q + score accums + softmax
// state) — give the RA the full budget rather than spill.
// ---------------------------------------------------------------------------
__global__ __launch_bounds__(128, 1)
void attn_kernel(const __bf16* __restrict__ qg,
                 const __bf16* __restrict__ kg,
                 const __bf16* __restrict__ vtg,
                 float* __restrict__ out)
{
    __shared__ __bf16 lds_p[4][32 * 32];  // 2 KB per wave, 4 waves/block

    const int lane  = threadIdx.x & 31;
    const int wslot = threadIdx.x >> 5;
    const int wid   = (blockIdx.x * blockDim.x + threadIdx.x) >> 5;
    const int b     = wid / (S_ / 32);
    const int q0    = (wid % (S_ / 32)) * 32;

    const int m = lane & 15;
    const int g = lane >> 4;

    // Q: two row-halves x two 32-wide K chunks, already scaled by 1/sqrt(HD).
    v16bf aq[2][2];
    #pragma unroll
    for (int h = 0; h < 2; ++h) {
        const __bf16* qrow = qg + (size_t)(b * S_ + q0 + 16 * h + m) * HD_;
        aq[h][0] = load_op16(qrow, g);
        aq[h][1] = load_op16(qrow + 32, g);
    }

    v8f o[2][4];
    float mrow[2][8], lrow[2][8];
    #pragma unroll
    for (int h = 0; h < 2; ++h) {
        #pragma unroll
        for (int t = 0; t < 4; ++t) o[h][t] = v8f_zero();
        #pragma unroll
        for (int r = 0; r < 8; ++r) { mrow[h][r] = -3.0e38f; lrow[h][r] = 0.0f; }
    }

    const int nblocks = (q0 >> 5) + 1;  // keys 0 .. q0+31
    for (int t = 0; t < nblocks; ++t) {
        const int k0 = t << 5;

        // Prefetch next key block's K and V rows into cache.
        if (t + 1 < nblocks) {
            const int kn = k0 + 32;
            __builtin_prefetch(kg + (size_t)(b * S_ + kn + m + 16 * g) * HD_, 0, 1);
            __builtin_prefetch(vtg + (size_t)(b * HD_ + m + 16 * g) * S_ + kn, 0, 1);
        }

        // ---- scores: sc[h][i] = Qh(16x64) . K^T subtile (64x16) ----
        v8f sc[2][2];
        #pragma unroll
        for (int i = 0; i < 2; ++i) {
            const int nk = k0 + i * 16 + m;  // max = q0+31 <= S-1
            const __bf16* krow = kg + (size_t)(b * S_ + nk) * HD_;
            const v16bf bk0 = load_op16(krow, g);
            const v16bf bk1 = load_op16(krow + 32, g);
            #pragma unroll
            for (int h = 0; h < 2; ++h) {
                sc[h][i] = v8f_zero();
                sc[h][i] = wmma_bf16(aq[h][0], bk0, sc[h][i]);
                sc[h][i] = wmma_bf16(aq[h][1], bk1, sc[h][i]);
            }
        }

        // ---- causal mask + online softmax (16-lane butterflies) ----
        #pragma unroll
        for (int h = 0; h < 2; ++h) {
            #pragma unroll
            for (int r = 0; r < 8; ++r) {
                const int mq = q0 + 16 * h + r + 8 * g;
                float s0v = (k0 + m)      <= mq ? sc[h][0][r] : -1.0e30f;
                float s1v = (k0 + 16 + m) <= mq ? sc[h][1][r] : -1.0e30f;
                float mx = fmaxf(s0v, s1v);
                #pragma unroll
                for (int off = 8; off; off >>= 1)
                    mx = fmaxf(mx, __shfl_xor(mx, off, 16));
                const float mnew  = fmaxf(mrow[h][r], mx);
                const float alpha = __expf(mrow[h][r] - mnew);
                const float p0 = __expf(s0v - mnew);
                const float p1 = __expf(s1v - mnew);
                float ps = p0 + p1;
                #pragma unroll
                for (int off = 8; off; off >>= 1)
                    ps += __shfl_xor(ps, off, 16);
                lrow[h][r] = lrow[h][r] * alpha + ps;
                mrow[h][r] = mnew;
                #pragma unroll
                for (int tt = 0; tt < 4; ++tt) o[h][tt][r] *= alpha;

                const int row = 16 * h + r + 8 * g;
                lds_p[wslot][row * 32 + m]      = (__bf16)p0;
                lds_p[wslot][row * 32 + 16 + m] = (__bf16)p1;
            }
        }

        // ---- reload P in A layout (same-wave DS: in-order, no barrier) ----
        v16bf pa[2];
        #pragma unroll
        for (int h = 0; h < 2; ++h)
            pa[h] = load_op16(&lds_p[wslot][(16 * h + m) * 32], g);

        // ---- O(32x64) += P(32x32) . V(32x64), V read from [B,HD,S] ----
        #pragma unroll
        for (int tt = 0; tt < 4; ++tt) {
            const int nh = tt * 16 + m;
            const __bf16* vrow = vtg + (size_t)(b * HD_ + nh) * S_;
            const v16bf bv = load_op16(vrow + k0, g);
            #pragma unroll
            for (int h = 0; h < 2; ++h)
                o[h][tt] = wmma_bf16(pa[h], bv, o[h][tt]);
        }
    }

    // ---- normalize by row sum and store f32 ----
    #pragma unroll
    for (int h = 0; h < 2; ++h) {
        #pragma unroll
        for (int tt = 0; tt < 4; ++tt) {
            const int n = tt * 16 + m;
            #pragma unroll
            for (int r = 0; r < 8; ++r) {
                const int row = 16 * h + r + 8 * g;
                out[(size_t)(b * S_ + q0 + row) * HD_ + n] = o[h][tt][r] / lrow[h][r];
            }
        }
    }
}

// ---------------------------------------------------------------------------
extern "C" void kernel_launch(void* const* d_in, const int* in_sizes, int n_in,
                              void* d_out, int out_size, void* d_ws, size_t ws_size,
                              hipStream_t stream) {
    const float* x  = (const float*)d_in[0];
    const float* wq = (const float*)d_in[1];
    const float* bq = (const float*)d_in[2];
    const float* wk = (const float*)d_in[3];
    const float* bk = (const float*)d_in[4];
    const float* wv = (const float*)d_in[5];
    const float* bv = (const float*)d_in[6];
    float* out = (float*)d_out;

    const size_t elems = (size_t)B_ * S_ * HD_;       // 1,048,576 per matrix
    const size_t wswel = (size_t)3 * (D_ / 32) * 4 * 32 * 16;  // 196,608
    __bf16* qws  = (__bf16*)d_ws;
    __bf16* kws  = qws + elems;
    __bf16* vtws = kws + elems;
    __bf16* wsw  = vtws + elems;   // total ~6.4 MB of workspace

    // Kernel 0: weight convert+swizzle, 196608 threads.
    wswizzle_kernel<<<(int)(wswel / 256), 256, 0, stream>>>(wq, wk, wv, wsw);

    // Kernel 1: one wave per 16-row tile, fused q/k/v: 1024 waves.
    const int proj_waves = B_ * S_ / 16;
    qkv_proj_kernel<<<proj_waves * 32 / 256, 256, 0, stream>>>(
        x, wsw, bq, bk, bv, qws, kws, vtws);

    // Kernel 2: one wave per 32-query tile: 512 waves, 4 waves/block.
    const int attn_waves = B_ * S_ / 32;
    attn_kernel<<<attn_waves * 32 / 128, 128, 0, stream>>>(qws, kws, vtws, out);
}
